// SmartCNN_64089501991284
// MI455X (gfx1250) — matrix operand order repaired and checked by reading, not compiled, not run on hardware
//
#include <hip/hip_runtime.h>

// CDNA5 / gfx1250, wave32.
typedef _Float16 half8   __attribute__((ext_vector_type(8)));
typedef _Float16 v16h    __attribute__((ext_vector_type(16)));
typedef float    v8f     __attribute__((ext_vector_type(8)));
typedef float    float4v __attribute__((ext_vector_type(4)));

#define KDIM 192            // 16 positions * 12 channels = 6 WMMA K-steps
#define NDIM 96             // outputs per board
#define WAVES_PER_WG 4
#define BOARDS_PER_WAVE 16
#define TILES_PER_WAVE 4
#define BOARDS_PER_WG (WAVES_PER_WG * BOARDS_PER_WAVE * TILES_PER_WAVE)  // 256

static __device__ inline v16h cat16(half8 lo, half8 hi) {
    v16h r;
#pragma unroll
    for (int t = 0; t < 8; ++t) { r[t] = lo[t]; r[t + 8] = hi[t]; }
    return r;
}

__global__ __launch_bounds__(128) void smartcnn_wmma_kernel(
    const float* __restrict__ x,    // [B,4,4]
    const float* __restrict__ W0,   // [4,12,1,2] OIHW
    const float* __restrict__ b0,   // [4]
    const float* __restrict__ W1,   // [4,12,2,1] OIHW
    float* __restrict__ out,        // [B,96]
    int nB)
{
    // B-operand (weights), transposed: row n (output col), K contiguous. 36 KB
    __shared__ __align__(16) _Float16 sW[NDIM][KDIM];
    // Per-wave A tiles; region is reused as the 16x96 f32 D buffer.     24 KB
    __shared__ __align__(16) _Float16 sA[WAVES_PER_WG][BOARDS_PER_WAVE][KDIM];
    __shared__ float sb0[4];

    const int tid  = threadIdx.x;
    const int wave = tid >> 5;
    const int lane = tid & 31;
    const int lrow = lane & 15;   // M row (A) / N column (B,D) / board id
    const int hi   = lane >> 4;   // half-wave selector

    if (tid < 4) sb0[tid] = b0[tid];

    // ---- Build Wmat^T in LDS (threads 0..95, one output row each) ----------
    if (tid < NDIM) {
        const int n = tid;
        _Float16* row = &sW[n][0];
        half8 z = {};
#pragma unroll
        for (int t = 0; t < KDIM / 8; ++t) *(half8*)(row + t * 8) = z;

        if (n < 48) {               // zh = relu(conv1x2(p, W0) + b0)
            const int o = n / 12, rem = n % 12, r = rem / 3, cc = rem % 3;
            const int p0 = r * 4 + cc, p1 = r * 4 + cc + 1;
#pragma unroll
            for (int c = 0; c < 12; ++c) {
                row[p0 * 12 + c] = (_Float16)W0[o * 24 + c * 2 + 0];
                row[p1 * 12 + c] = (_Float16)W0[o * 24 + c * 2 + 1];
            }
        } else {                    // zv^T = relu(conv2x1(p, W1)), transposed
            const int m = n - 48;
            const int o = m / 12, rem = m % 12, r = rem / 3, cc = rem % 3;
            const int p0 = cc * 4 + r, p1 = (cc + 1) * 4 + r;
#pragma unroll
            for (int c = 0; c < 12; ++c) {
                row[p0 * 12 + c] = (_Float16)W1[o * 24 + c * 2 + 0];
                row[p1 * 12 + c] = (_Float16)W1[o * 24 + c * 2 + 1];
            }
        }
    }

    __syncthreads();   // only sW/sb0 need WG-wide visibility

    _Float16* aRow = &sA[wave][lrow][0];        // build + A-fragment row
    float*    sD   = (float*)&sA[wave][0][0];   // 16x96 f32 D buffer (aliases sA)

    const int waveBase =
        blockIdx.x * BOARDS_PER_WG + wave * (BOARDS_PER_WAVE * TILES_PER_WAVE);

    for (int t = 0; t < TILES_PER_WAVE; ++t) {
        const int tileBase = waveBase + t * BOARDS_PER_WAVE;
        int bglobal = tileBase + lrow;
        if (bglobal >= nB) bglobal = nB - 1;    // clamp loads; stores guarded

        // prefetch next tile's board row while this tile computes
        {
            int pb = bglobal + BOARDS_PER_WAVE;
            if (pb >= nB) pb = nB - 1;
            __builtin_prefetch(&x[(size_t)pb * 16], 0, 1);
        }

        // ---- zero + one-hot scatter into the A tile ------------------------
        {
            half8 z = {};
            _Float16* zdst = aRow + hi * 96;
#pragma unroll
            for (int q = 0; q < 12; ++q) *(half8*)(zdst + q * 8) = z;
        }

        const size_t xb = (size_t)bglobal * 16;
        float c0 = x[xb + 0], c1 = x[xb + 3], c2 = x[xb + 12], c3 = x[xb + 15];
        int ixm = 0; float best = c0;           // first-max wins == jnp.argmax
        if (c1 > best) { best = c1; ixm = 1; }
        if (c2 > best) { best = c2; ixm = 2; }
        if (c3 > best) { best = c3; ixm = 3; }
        const bool fv = (ixm >> 1) & 1;
        const bool fh = ixm & 1;
        const int  fm = (fv ? 12 : 0) | (fh ? 3 : 0);  // flip == XOR on pos
        const int  cMask = fv ? 1 : 0;

#pragma unroll
        for (int p = 0; p < 8; ++p) {
            const int pos = hi * 8 + p;
            const float v = x[xb + (pos ^ fm)];
            int e = (int)(__float_as_uint(v) >> 23) - 127;  // exact log2
            if (e < 0) e = 0;                               // v==0 -> class 0
            int cc2 = e + 1;                                // in 1..11
            if (fv && cc2 == 1) cc2 = 0;                    // chan swap 0<->1
            if (fh) cc2 = (cc2 == 2) ? 3 : (cc2 == 3 ? 2 : cc2);  // 2<->3
            aRow[pos * 12 + cMask] = (_Float16)1.0f;
            aRow[pos * 12 + cc2]   = (_Float16)1.0f;
        }

        // ---- A fragments (same-wave LDS RAW is in-order) -------------------
        // 16-bit A 16x32: lanes 0-15 hold K 0..7/16..23, lanes 16-31 K 8..15/24..31
        v16h A[6];
#pragma unroll
        for (int ks = 0; ks < 6; ++ks) {
            const int kb = ks * 32;
            half8 lo8 = *(const half8*)(aRow + kb + hi * 8);
            half8 hi8 = *(const half8*)(aRow + kb + 16 + hi * 8);
            A[ks] = cat16(lo8, hi8);
        }

        // ---- GEMM: 6 N-tiles x 6 K-steps = 36 WMMA; D -> LDS ---------------
        float* dLane = sD + (hi * 8) * 96 + lrow;  // + q*96 + nt*16 (immediates)
#pragma unroll
        for (int nt = 0; nt < 6; ++nt) {
            const int n = nt * 16 + lrow;
            const _Float16* bRow = &sW[n][0];
            v8f C = {};
#pragma unroll
            for (int ks = 0; ks < 6; ++ks) {
                // 16-bit B 32x16: lane = N col; lanes 0-15 K 0..7/16..23, lanes 16-31 K 8..15/24..31
                const int kb = ks * 32 + hi * 8;
                half8 blo = *(const half8*)(bRow + kb);
                half8 bhi = *(const half8*)(bRow + kb + 16);
                v16h Bm = cat16(blo, bhi);
                C = __builtin_amdgcn_wmma_f32_16x16x32_f16(
                        /*neg_a=*/false, A[ks], /*neg_b=*/false, Bm,
                        /*c_mod=*/(short)0, C, /*reuse_a=*/false, /*reuse_b=*/false);
            }
            const float bias = (n < 48) ? sb0[n / 12] : 0.0f;
#pragma unroll
            for (int q = 0; q < 8; ++q) {
                dLane[q * 96 + nt * 16] = fmaxf(C[q] + bias, 0.0f);
            }
        }

        // ---- coalesced write-out: 12 x b128 per lane -----------------------
        // lane covers board lrow, floats [hi*48, hi*48+48)
        if (tileBase + lrow < nB) {
            const float* dsrc = sD + lrow * 96 + hi * 48;
            float* gdst = out + (size_t)(tileBase + lrow) * 96 + hi * 48;
#pragma unroll
            for (int q = 0; q < 12; ++q) {
                *(float4v*)(gdst + q * 4) = *(const float4v*)(dsrc + q * 4);
            }
        }
        // next tile's sA zeroing is ordered after these ds_loads (in-order DS)
    }
}

extern "C" void kernel_launch(void* const* d_in, const int* in_sizes, int n_in,
                              void* d_out, int out_size, void* d_ws, size_t ws_size,
                              hipStream_t stream) {
    const float* x  = (const float*)d_in[0];
    const float* W0 = (const float*)d_in[1];
    const float* b0 = (const float*)d_in[2];
    const float* W1 = (const float*)d_in[3];
    float* out = (float*)d_out;

    const int nB = in_sizes[0] / 16;
    const int blocks = (nB + BOARDS_PER_WG - 1) / BOARDS_PER_WG;
    smartcnn_wmma_kernel<<<blocks, 128, 0, stream>>>(x, W0, b0, W1, out, nB);
}
